// SOMVAE_34548716929291
// MI455X (gfx1250) — matrix operand order repaired and checked by reading, not compiled, not run on hardware
//
#include <hip/hip_runtime.h>
#include <hip/hip_bf16.h>

// ---------------------------------------------------------------------------
// SOM-VAE forward for MI455X (gfx1250): split-bf16 WMMA GEMM pipeline, v2.
// Activations kept as bf16 hi/lo pairs in memory; inner loop is pure
// global_load_b128 + v_wmma_f32_16x16x32_bf16 (no LDS, no conversions).
// ---------------------------------------------------------------------------

typedef __attribute__((ext_vector_type(8)))  float          v8f;
typedef __attribute__((ext_vector_type(16))) __bf16         v16bf;
typedef __attribute__((ext_vector_type(8)))  __bf16         v8bf;

#define BDIM   16384
#define SDIM   256
#define CDIM   8
#define DIN    2048      // S*C
#define HIDD   512
#define LATD   64

__device__ __forceinline__ unsigned short somvae_f2bf(float f) {
    __bf16 h = (__bf16)f;
    return __builtin_bit_cast(unsigned short, h);
}

// ---------------------------------------------------------------------------
// Weight pre-transform: W fp32 [K,N]  ->  Wt_hi/Wt_lo bf16 [N,K] (K contiguous)
// ---------------------------------------------------------------------------
__global__ void somvae_wsplit_k(const float* __restrict__ W,
                                unsigned short* __restrict__ hi,
                                unsigned short* __restrict__ lo,
                                int K, int N) {
    size_t idx = (size_t)blockIdx.x * blockDim.x + threadIdx.x;
    if (idx >= (size_t)K * N) return;
    int n = (int)(idx / K);
    int k = (int)(idx % K);
    float v = W[(size_t)k * N + n];
    __bf16 h = (__bf16)v;
    hi[idx] = __builtin_bit_cast(unsigned short, h);
    lo[idx] = somvae_f2bf(v - (float)h);
}

// ---------------------------------------------------------------------------
// x pre-split: fp32 [B,DIN] -> hi/lo bf16 [B,DIN] (same row-major layout)
// ---------------------------------------------------------------------------
__global__ void somvae_xsplit_k(const float* __restrict__ x,
                                unsigned short* __restrict__ hi,
                                unsigned short* __restrict__ lo,
                                size_t n4) {
    size_t i = (size_t)blockIdx.x * blockDim.x + threadIdx.x;
    if (i >= n4) return;
    float4 v = ((const float4*)x)[i];
    ushort4 h, l;
    {
        __bf16 a = (__bf16)v.x; h.x = __builtin_bit_cast(unsigned short, a); l.x = somvae_f2bf(v.x - (float)a);
    }
    {
        __bf16 a = (__bf16)v.y; h.y = __builtin_bit_cast(unsigned short, a); l.y = somvae_f2bf(v.y - (float)a);
    }
    {
        __bf16 a = (__bf16)v.z; h.z = __builtin_bit_cast(unsigned short, a); l.z = somvae_f2bf(v.z - (float)a);
    }
    {
        __bf16 a = (__bf16)v.w; h.w = __builtin_bit_cast(unsigned short, a); l.w = somvae_f2bf(v.w - (float)a);
    }
    ((ushort4*)hi)[i] = h;
    ((ushort4*)lo)[i] = l;
}

// ---------------------------------------------------------------------------
// GEMM: C[M,N] = act( A[M,K] @ W[K,N] + bias[N] )
// A given as split bf16 hi/lo [M,K]; W as split bf16 transposed [N,K].
// Block = 128 threads (4 waves); wave tile 16 x (16*NT); block tile 16 x 64*NT.
// ACT: 0 none, 1 relu, 2 tanh.
// EPI: 0 = fp32 store, 1 = split-bf16 store, 2 = both, 3 = fused final
//      (reads x & x_q, writes x_q + x_e, accumulates both MSE sums).
// ---------------------------------------------------------------------------
template <int ACT, int EPI, int NT, int KDIM, int NDIM>
__global__ __launch_bounds__(128)
void somvae_gemm_k(const unsigned short* __restrict__ Ah,
                   const unsigned short* __restrict__ Al,
                   const unsigned short* __restrict__ Wh,
                   const unsigned short* __restrict__ Wl,
                   const float* __restrict__ bias,
                   float* __restrict__ Cf,
                   unsigned short* __restrict__ Chi,
                   unsigned short* __restrict__ Clo,
                   const float* __restrict__ Xref,   // EPI==3
                   float* __restrict__ Out,          // EPI==3 (holds x_q, gets x_q+x_e)
                   double* __restrict__ acc)         // EPI==3
{
    const int m0   = blockIdx.x * 16;
    const int wave = threadIdx.x >> 5;
    const int lane = threadIdx.x & 31;
    const int row  = lane & 15;
    const int half = lane >> 4;
    const int nbase = blockIdx.y * (64 * NT) + wave * (16 * NT);

    v8f cacc[NT];
#pragma unroll
    for (int t = 0; t < NT; ++t) cacc[t] = (v8f){};

    const size_t arow = (size_t)(m0 + row) * KDIM;

    for (int k0 = 0; k0 < KDIM; k0 += 32) {
        // A fragments (16x32 bf16, this code's contiguous k order): lane L: M = L%16,
        // elements 0..15 -> K = k0 + 16*half + j (contiguous, matches B)
        v8bf ah0 = *(const v8bf*)&Ah[arow + k0 + 16 * half];
        v8bf ah1 = *(const v8bf*)&Ah[arow + k0 + 16 * half + 8];
        v8bf al0 = *(const v8bf*)&Al[arow + k0 + 16 * half];
        v8bf al1 = *(const v8bf*)&Al[arow + k0 + 16 * half + 8];
        v16bf ahv = __builtin_shufflevector(ah0, ah1, 0, 1, 2, 3, 4, 5, 6, 7,
                                            8, 9, 10, 11, 12, 13, 14, 15);
        v16bf alv = __builtin_shufflevector(al0, al1, 0, 1, 2, 3, 4, 5, 6, 7,
                                            8, 9, 10, 11, 12, 13, 14, 15);
#pragma unroll
        for (int t = 0; t < NT; ++t) {
            // B fragment: lane L: N = L%16, K = k0 + 16*half + j (contiguous)
            const size_t boff = (size_t)(nbase + t * 16 + row) * KDIM
                              + (size_t)(k0 + 16 * half);
            v16bf bh = *(const v16bf*)&Wh[boff];
            v16bf bl = *(const v16bf*)&Wl[boff];
            // split-precision: ah*bh + ah*bl + al*bh
            cacc[t] = __builtin_amdgcn_wmma_f32_16x16x32_bf16(
                false, ahv, false, bh, (short)0, cacc[t], false, false);
            cacc[t] = __builtin_amdgcn_wmma_f32_16x16x32_bf16(
                false, ahv, false, bl, (short)0, cacc[t], false, false);
            cacc[t] = __builtin_amdgcn_wmma_f32_16x16x32_bf16(
                false, alv, false, bh, (short)0, cacc[t], false, false);
        }
    }

    // epilogue: C layout lane L -> N = nbase + t*16 + L%16, M = m0 + r + 8*half
    float se = 0.f, sq = 0.f;
#pragma unroll
    for (int t = 0; t < NT; ++t) {
        const int   n  = nbase + t * 16 + row;
        const float bn = bias[n];
#pragma unroll
        for (int r = 0; r < 8; ++r) {
            int    m = m0 + r + 8 * half;
            float  v = cacc[t][r] + bn;
            if (ACT == 1) v = fmaxf(v, 0.f);
            if (ACT == 2) v = tanhf(v);
            size_t o = (size_t)m * NDIM + n;
            if (EPI == 0 || EPI == 2) Cf[o] = v;
            if (EPI == 1 || EPI == 2) {
                __bf16 h = (__bf16)v;
                Chi[o] = __builtin_bit_cast(unsigned short, h);
                Clo[o] = somvae_f2bf(v - (float)h);
            }
            if (EPI == 3) {
                float xq = Out[o];
                float xv = Xref[o];
                Out[o] = xq + v;
                float de = xv - v;  se += de * de;
                float dq = xv - xq; sq += dq * dq;
            }
        }
    }
    if (EPI == 3) {
#pragma unroll
        for (int off = 16; off > 0; off >>= 1) {
            se += __shfl_xor(se, off);
            sq += __shfl_xor(sq, off);
        }
        if (lane == 0) {
            atomicAdd(&acc[0], (double)se);   // sum (x - x_e)^2
            atomicAdd(&acc[1], (double)sq);   // sum (x - x_q)^2
        }
    }
}

// ---------------------------------------------------------------------------
// Codebook / SOM kernel: per-row argmin over 9 codes, split-bf16 z_q emit,
// commit + SOM-neighbor loss partials (masked neighbors -> z_e^2 term).
// ---------------------------------------------------------------------------
__global__ __launch_bounds__(256)
void somvae_som_k(const float* __restrict__ ze,
                  const float* __restrict__ emb,   // [9,64]
                  unsigned short* __restrict__ zqh,
                  unsigned short* __restrict__ zql,
                  double* __restrict__ acc, int Bn)
{
    int b = blockIdx.x * blockDim.x + threadIdx.x;
    float commit = 0.f, som = 0.f;
    if (b < Bn) {
        float z[64];
        const float4* zp = (const float4*)&ze[(size_t)b * 64];
#pragma unroll
        for (int i = 0; i < 16; ++i) ((float4*)z)[i] = zp[i];

        float best  = 3.4e38f;
        int   kbest = 0;
        for (int j = 0; j < 9; ++j) {
            float d = 0.f;
#pragma unroll
            for (int i = 0; i < 64; ++i) {
                float t = z[i] - emb[j * 64 + i];
                d += t * t;
            }
            if (d < best) { best = d; kbest = j; }   // first-min like jnp.argmin
        }
        int k1 = kbest / 3, k2 = kbest % 3;

#pragma unroll
        for (int i = 0; i < 64; ++i) {
            float  q = emb[kbest * 64 + i];
            __bf16 h = (__bf16)q;
            zqh[(size_t)b * 64 + i] = __builtin_bit_cast(unsigned short, h);
            zql[(size_t)b * 64 + i] = somvae_f2bf(q - (float)h);
            float c = z[i] - q;
            commit += c * c;
            som    += c * c;                 // z_q itself is a "neighbor"
        }
        int  nb[4];
        bool ok[4];
        ok[0] = (k1 < 2); nb[0] = (k1 + 1) * 3 + k2;   // up
        ok[1] = (k1 > 0); nb[1] = (k1 - 1) * 3 + k2;   // down
        ok[2] = (k2 < 2); nb[2] = k1 * 3 + (k2 + 1);   // right
        ok[3] = (k2 > 0); nb[3] = k1 * 3 + (k2 - 1);   // left
        for (int t = 0; t < 4; ++t) {
            if (ok[t]) {
#pragma unroll
                for (int i = 0; i < 64; ++i) {
                    float c = z[i] - emb[nb[t] * 64 + i];
                    som += c * c;
                }
            } else {
#pragma unroll
                for (int i = 0; i < 64; ++i) som += z[i] * z[i];   // (z_e - 0)^2
            }
        }
    }
#pragma unroll
    for (int off = 16; off > 0; off >>= 1) {
        commit += __shfl_xor(commit, off);
        som    += __shfl_xor(som, off);
    }
    if ((threadIdx.x & 31) == 0) {
        atomicAdd(&acc[2], (double)commit);
        atomicAdd(&acc[3], (double)som);
    }
}

__global__ void somvae_zero_acc_k(double* a) {
    if (threadIdx.x < 8) a[threadIdx.x] = 0.0;
}

__global__ void somvae_finalize_k(const double* __restrict__ a,
                                  float* __restrict__ out) {
    if (threadIdx.x == 0) {
        double mse    = (a[0] + a[1]) / ((double)BDIM * SDIM * CDIM);
        double commit = a[2] / ((double)BDIM * LATD);
        double som    = a[3] / ((double)BDIM * 5 * LATD);
        out[0] = (float)(mse + commit + som);
    }
}

// ---------------------------------------------------------------------------
// Host orchestration
// ---------------------------------------------------------------------------
static inline size_t somvae_align256(size_t x) { return (x + 255) & ~(size_t)255; }

extern "C" void kernel_launch(void* const* d_in, const int* in_sizes, int n_in,
                              void* d_out, int out_size, void* d_ws, size_t ws_size,
                              hipStream_t stream) {
    (void)in_sizes; (void)n_in; (void)out_size; (void)ws_size;

    const float* x   = (const float*)d_in[0];
    const float* emb = (const float*)d_in[20];
    float*       out = (float*)d_out;
    char*        ws  = (char*)d_ws;

    // weight matrices: {K,N}; weight input index (bias at widx+1)
    const int KN[9][2] = {{DIN, HIDD}, {HIDD, HIDD}, {HIDD, LATD},
                          {LATD, HIDD}, {HIDD, HIDD}, {HIDD, DIN},
                          {LATD, HIDD}, {HIDD, HIDD}, {HIDD, DIN}};
    const int widx[9]  = {2, 4, 6, 8, 10, 12, 14, 16, 18};

    size_t off = 0;
    double* acc = (double*)(ws + off); off += 256;

    unsigned short* whi[9];
    unsigned short* wlo[9];
    for (int i = 0; i < 9; ++i) {
        size_t e = (size_t)KN[i][0] * KN[i][1];
        whi[i] = (unsigned short*)(ws + off); off += somvae_align256(e * 2);
        wlo[i] = (unsigned short*)(ws + off); off += somvae_align256(e * 2);
    }
    unsigned short* xhi = (unsigned short*)(ws + off); off += (size_t)BDIM * DIN * 2;
    unsigned short* xlo = (unsigned short*)(ws + off); off += (size_t)BDIM * DIN * 2;
    unsigned short* bAh = (unsigned short*)(ws + off); off += (size_t)BDIM * HIDD * 2;
    unsigned short* bAl = (unsigned short*)(ws + off); off += (size_t)BDIM * HIDD * 2;
    unsigned short* bBh = (unsigned short*)(ws + off); off += (size_t)BDIM * HIDD * 2;
    unsigned short* bBl = (unsigned short*)(ws + off); off += (size_t)BDIM * HIDD * 2;
    float*          z_e = (float*)(ws + off);          off += (size_t)BDIM * LATD * 4;
    unsigned short* zeh = (unsigned short*)(ws + off); off += (size_t)BDIM * LATD * 2;
    unsigned short* zel = (unsigned short*)(ws + off); off += (size_t)BDIM * LATD * 2;
    unsigned short* zqh = (unsigned short*)(ws + off); off += (size_t)BDIM * LATD * 2;
    unsigned short* zql = (unsigned short*)(ws + off); off += (size_t)BDIM * LATD * 2;

    // 0) zero loss accumulators
    somvae_zero_acc_k<<<dim3(1), dim3(32), 0, stream>>>(acc);

    // 1) split weights (transposed) and x
    for (int i = 0; i < 9; ++i) {
        size_t e = (size_t)KN[i][0] * KN[i][1];
        int blocks = (int)((e + 255) / 256);
        somvae_wsplit_k<<<dim3(blocks), dim3(256), 0, stream>>>(
            (const float*)d_in[widx[i]], whi[i], wlo[i], KN[i][0], KN[i][1]);
    }
    {
        size_t n4 = (size_t)BDIM * DIN / 4;
        somvae_xsplit_k<<<dim3((unsigned)((n4 + 255) / 256)), dim3(256), 0, stream>>>(
            x, xhi, xlo, n4);
    }

    const dim3 gblk(128);
    const int  mg = BDIM / 16;   // 1024

    // 2) encoder
    somvae_gemm_k<1, 1, 2, DIN, HIDD><<<dim3(mg, HIDD / 128), gblk, 0, stream>>>(
        xhi, xlo, whi[0], wlo[0], (const float*)d_in[3],
        nullptr, bAh, bAl, nullptr, nullptr, nullptr);
    somvae_gemm_k<1, 1, 2, HIDD, HIDD><<<dim3(mg, HIDD / 128), gblk, 0, stream>>>(
        bAh, bAl, whi[1], wlo[1], (const float*)d_in[5],
        nullptr, bBh, bBl, nullptr, nullptr, nullptr);
    somvae_gemm_k<2, 2, 1, HIDD, LATD><<<dim3(mg, 1), gblk, 0, stream>>>(
        bBh, bBl, whi[2], wlo[2], (const float*)d_in[7],
        z_e, zeh, zel, nullptr, nullptr, nullptr);

    // 3) codebook argmin + commit/SOM loss partials (emits split z_q)
    somvae_som_k<<<dim3(BDIM / 256), dim3(256), 0, stream>>>(z_e, emb, zqh, zql, acc, BDIM);

    // 4) decoder-q (x_q -> d_out[1:])
    somvae_gemm_k<1, 1, 2, LATD, HIDD><<<dim3(mg, HIDD / 128), gblk, 0, stream>>>(
        zqh, zql, whi[3], wlo[3], (const float*)d_in[9],
        nullptr, bAh, bAl, nullptr, nullptr, nullptr);
    somvae_gemm_k<1, 1, 2, HIDD, HIDD><<<dim3(mg, HIDD / 128), gblk, 0, stream>>>(
        bAh, bAl, whi[4], wlo[4], (const float*)d_in[11],
        nullptr, bBh, bBl, nullptr, nullptr, nullptr);
    somvae_gemm_k<0, 0, 2, HIDD, DIN><<<dim3(mg, DIN / 128), gblk, 0, stream>>>(
        bBh, bBl, whi[5], wlo[5], (const float*)d_in[13],
        out + 1, nullptr, nullptr, nullptr, nullptr, nullptr);

    // 5) decoder-e, fused with x_q + x_e output and MSE accumulation
    somvae_gemm_k<1, 1, 2, LATD, HIDD><<<dim3(mg, HIDD / 128), gblk, 0, stream>>>(
        zeh, zel, whi[6], wlo[6], (const float*)d_in[15],
        nullptr, bAh, bAl, nullptr, nullptr, nullptr);
    somvae_gemm_k<1, 1, 2, HIDD, HIDD><<<dim3(mg, HIDD / 128), gblk, 0, stream>>>(
        bAh, bAl, whi[7], wlo[7], (const float*)d_in[17],
        nullptr, bBh, bBl, nullptr, nullptr, nullptr);
    somvae_gemm_k<0, 3, 2, HIDD, DIN><<<dim3(mg, DIN / 128), gblk, 0, stream>>>(
        bBh, bBl, whi[8], wlo[8], (const float*)d_in[19],
        nullptr, nullptr, nullptr, x, out + 1, acc);

    // 6) scalar loss
    somvae_finalize_k<<<dim3(1), dim3(32), 0, stream>>>(acc, out);
}
